// PropagatorWithForest_16028817949143
// MI455X (gfx1250) — matrix operand order repaired and checked by reading, not compiled, not run on hardware
//
#include <hip/hip_runtime.h>

typedef __bf16 bf16_t;
typedef __attribute__((ext_vector_type(16))) __bf16 v16bf;
typedef __attribute__((ext_vector_type(8)))  float  v8f;

#define NSEG 16384
#define DIM 128
#define TLEN 50
#define ROWS_PER_WG 16
#define ZSB 136        // bf16 row stride (128 + 8 pad)

__device__ __forceinline__ bf16_t f2bf(float f) { return (bf16_t)f; }
__device__ __forceinline__ float  bf2f(bf16_t h) { return (float)h; }

struct b32x8 { uint4 a, b; };
// A-style fragment: elements 0-7 at p[0..7], elements 8-15 at p[16..23]
// (K groups {kbase..kbase+7} and {16+kbase..16+kbase+7}).
__device__ __forceinline__ v16bf ld_a16(const bf16_t* p) {
    b32x8 t;
    t.a = *(const uint4*)(p);
    t.b = *(const uint4*)(p + 16);
    return __builtin_bit_cast(v16bf, t);
}

// WG = 4 waves sharing 16 batch rows. Each wave owns 2 of the 8 output tiles
// and keeps its W slice (bf16 hi+lo) permanently in VGPRs. z state lives in
// ping-pong LDS buffers pre-split into bf16 hi/lo, so each step needs exactly
// ONE workgroup barrier and A-fragments load with zero conversion work.
// Step: z_new = alpha * (z @ W^T) + h via 3-term bf16-split WMMA.
__global__ void __launch_bounds__(128) prop_kernel(
    const float* __restrict__ feats,   // (N, D) f32
    const float* __restrict__ attn,    // (N, N) f32
    const float* __restrict__ projw,   // (D, D) f32
    const int*   __restrict__ paths,   // (B, T) int32
    float*       __restrict__ out)     // [N*D sums | N counts]
{
    __shared__ bf16_t zhi0[ROWS_PER_WG * ZSB], zlo0[ROWS_PER_WG * ZSB];
    __shared__ bf16_t zhi1[ROWS_PER_WG * ZSB], zlo1[ROWS_PER_WG * ZSB];
    __shared__ __align__(16) float ab0[ROWS_PER_WG], ab1[ROWS_PER_WG];
    __shared__ __align__(16) int   sb0[ROWS_PER_WG], sb1[ROWS_PER_WG];

    const int tid  = threadIdx.x;
    const int wave = tid >> 5;
    const int lane = tid & 31;

    const int arow  = lane & 15;       // A: lane -> row M
    const int half  = lane >> 4;       // lane half -> K sub-group
    const int kbase = half * 8;
    const int ncol  = lane & 15;       // B/C/D: lane -> column N
    const int rowbase = blockIdx.x * ROWS_PER_WG;
    const int tn0 = wave * 2;          // this wave's two output tiles

    // ---- Load this wave's W slice into registers (bf16 hi + residual lo) ----
    v16bf whi[2][4], wlo[2][4];
#pragma unroll
    for (int j = 0; j < 2; ++j) {
#pragma unroll
        for (int kc = 0; kc < 4; ++kc) {
            const float* wp = projw + ((tn0 + j) * 16 + ncol) * DIM + kc * 32 + half * 8;
            float4 a0 = *(const float4*)(wp);
            float4 a1 = *(const float4*)(wp + 4);
            float4 a2 = *(const float4*)(wp + 16);
            float4 a3 = *(const float4*)(wp + 20);
            float v[16] = { a0.x, a0.y, a0.z, a0.w,  a1.x, a1.y, a1.z, a1.w,
                            a2.x, a2.y, a2.z, a2.w,  a3.x, a3.y, a3.z, a3.w };
#pragma unroll
            for (int e = 0; e < 16; ++e) {
                bf16_t hi = f2bf(v[e]);
                whi[j][kc][e] = hi;
                wlo[j][kc][e] = f2bf(v[e] - bf2f(hi));
            }
        }
    }

    // ---- Prologue: z0 into buffer 0; alpha/seg for step 1 into ab0/sb0 ----
    const int r = rowbase + lane;      // meaningful for lane<16 of wave 0
    int seg0 = 0;
    if (wave == 0 && lane < 16) {
        seg0 = paths[r * TLEN + 0];
        sb1[lane] = seg0;              // temporary broadcast of seg0
    }
    __syncthreads();
    {
        int row = tid >> 3;            // 8 threads per row
        int c0  = (tid & 7) * 16;
        int seg = sb1[row];
#pragma unroll
        for (int g = 0; g < 4; ++g) {
            float4 v = *(const float4*)(feats + seg * DIM + c0 + g * 4);
            float f[4] = { v.x, v.y, v.z, v.w };
#pragma unroll
            for (int e = 0; e < 4; ++e) {
                bf16_t hi = f2bf(f[e]);
                int idx = row * ZSB + c0 + g * 4 + e;
                zhi0[idx] = hi;
                zlo0[idx] = f2bf(f[e] - bf2f(hi));
            }
        }
    }
    // Publish step-1 alpha/seg; preload step-2 alpha/seg into registers.
    float alphaNext = 0.0f;            // values for step t+1 (held by wave 0)
    int   segNext   = 0;
    if (wave == 0 && lane < 16) {
        int seg1 = paths[r * TLEN + 1];
        ab0[lane] = attn[seg0 * NSEG + seg1];
        sb0[lane] = seg1;
        int seg2 = paths[r * TLEN + 2];
        alphaNext = attn[seg1 * NSEG + seg2];
        segNext   = seg2;
    }
    __syncthreads();

    // ---- One scan step: read (szhi,szlo,aIn,sIn), write (dzhi,dzlo); publish
    //      step t+1 alpha/seg into (aOut,sOut); prefetch step t+2. One barrier.
    auto step = [&](int t,
                    const bf16_t* szhi, const bf16_t* szlo,
                    bf16_t* dzhi, bf16_t* dzlo,
                    const float* aIn, const int* sIn,
                    float* aOut, int* sOut) {
        // Publish t+1 (disjoint buffer from this step's readers).
        if (wave == 0 && lane < 16 && t < TLEN - 1) {
            aOut[lane] = alphaNext;
            sOut[lane] = segNext;
        }

        // A fragments: direct bf16 loads, no conversion.
        v16bf ahi[4], alo_[4];
#pragma unroll
        for (int kc = 0; kc < 4; ++kc) {
            const int off = arow * ZSB + kc * 32 + kbase;
            ahi[kc]  = ld_a16(szhi + off);
            alo_[kc] = ld_a16(szlo + off);
        }

        // Snapshot alpha/seg with vector LDS loads (rows half*8 .. half*8+7).
        const float4 a01 = *(const float4*)(aIn + half * 8);
        const float4 a23 = *(const float4*)(aIn + half * 8 + 4);
        const int4   s01 = *(const int4*)(sIn + half * 8);
        const int4   s23 = *(const int4*)(sIn + half * 8 + 4);
        float alpha_r[8] = { a01.x, a01.y, a01.z, a01.w,  a23.x, a23.y, a23.z, a23.w };
        int   seg_r[8]   = { s01.x, s01.y, s01.z, s01.w,  s23.x, s23.y, s23.z, s23.w };

        // Batched h gather (C/D layout) into registers.
        float hreg[16];
#pragma unroll
        for (int v = 0; v < 8; ++v) {
            hreg[v]     = feats[seg_r[v] * DIM + (tn0 + 0) * 16 + ncol];
            hreg[8 + v] = feats[seg_r[v] * DIM + (tn0 + 1) * 16 + ncol];
        }

        // Prefetch step t+2 path/alpha (hidden under compute).
        if (wave == 0 && lane < 16 && t < TLEN - 2) {
            int s2 = paths[r * TLEN + (t + 2)];
            alphaNext = attn[segNext * NSEG + s2];
            segNext   = s2;
        }

#pragma unroll
        for (int j = 0; j < 2; ++j) {
            v8f acc = {};
#pragma unroll
            for (int kc = 0; kc < 4; ++kc) {
                acc = __builtin_amdgcn_wmma_f32_16x16x32_bf16(
                          false, ahi[kc],  false, whi[j][kc], (short)0, acc, false, false);
                acc = __builtin_amdgcn_wmma_f32_16x16x32_bf16(
                          false, ahi[kc],  false, wlo[j][kc], (short)0, acc, false, false);
                acc = __builtin_amdgcn_wmma_f32_16x16x32_bf16(
                          false, alo_[kc], false, whi[j][kc], (short)0, acc, false, false);
            }
            // z_new = alpha*mm + h; split hi/lo into the destination buffer.
#pragma unroll
            for (int v = 0; v < 8; ++v) {
                float z = alpha_r[v] * acc[v] + hreg[j * 8 + v];
                bf16_t hi = f2bf(z);
                int idx = (v + half * 8) * ZSB + (tn0 + j) * 16 + ncol;
                dzhi[idx] = hi;
                dzlo[idx] = f2bf(z - bf2f(hi));
            }
        }
        __syncthreads();   // writes(t) visible before reads(t+1)
    };

    // 49 steps: step 1 (buf0 -> buf1), then 24 double-steps. Step 49 ends in buf1.
    step(1, zhi0, zlo0, zhi1, zlo1, ab0, sb0, ab1, sb1);
#pragma unroll 1
    for (int i = 0; i < 24; ++i) {
        step(2 + 2 * i, zhi1, zlo1, zhi0, zlo0, ab1, sb1, ab0, sb0);
        step(3 + 2 * i, zhi0, zlo0, zhi1, zlo1, ab0, sb0, ab1, sb1);
    }

    // ---- Segment-sum scatter: seg of step 49 is in sb0; z in zhi1/zlo1 ----
#pragma unroll 1
    for (int i = 0; i < 4; ++i) {
        int row = wave * 4 + i;
        int seg = sb0[row];
        int c = lane * 4;
        const bf16_t* ph = &zhi1[row * ZSB + c];
        const bf16_t* pl = &zlo1[row * ZSB + c];
        float* dst = out + seg * DIM + c;
#pragma unroll
        for (int e = 0; e < 4; ++e) {
            float z = bf2f(ph[e]) + bf2f(pl[e]);
            atomicAdd(dst + e, z);
        }
    }
    if (wave == 0 && lane < 16) atomicAdd(out + NSEG * DIM + sb0[lane], 1.0f);
}

__global__ void zero_kernel(float* p, int n) {
    int i = blockIdx.x * blockDim.x + threadIdx.x;
    if (i < n) p[i] = 0.0f;
}

// In-place mean: z_end = sums / max(counts,1) (0 where count==0); counts stay.
__global__ void finalize_kernel(float* out) {
    int seg = blockIdx.x;
    int c = threadIdx.x;
    float cnt = out[NSEG * DIM + seg];
    float val = out[seg * DIM + c];
    out[seg * DIM + c] = (cnt > 0.0f) ? val / fmaxf(cnt, 1.0f) : 0.0f;
}

extern "C" void kernel_launch(void* const* d_in, const int* in_sizes, int n_in,
                              void* d_out, int out_size, void* d_ws, size_t ws_size,
                              hipStream_t stream) {
    (void)in_sizes; (void)n_in; (void)d_ws; (void)ws_size;
    const float* feats = (const float*)d_in[0];   // (N, n_h, C) = (N, D)
    const float* attn  = (const float*)d_in[1];   // (N, N)
    const float* projw = (const float*)d_in[2];   // (D, D)
    const int*   paths = (const int*)d_in[3];     // (B, T)
    float* out = (float*)d_out;                   // [N*D | N]

    const int ntot = NSEG * DIM + NSEG;
    zero_kernel<<<(ntot + 255) / 256, 256, 0, stream>>>(out, ntot);

    prop_kernel<<<16384 / ROWS_PER_WG, 128, 0, stream>>>(feats, attn, projw, paths, out);

    finalize_kernel<<<NSEG, DIM, 0, stream>>>(out);
}
